// DendSeqNetSVHN2_45930380263777
// MI455X (gfx1250) — matrix operand order repaired and checked by reading, not compiled, not run on hardware
//
#include <hip/hip_runtime.h>
#include <hip/hip_bf16.h>

typedef __attribute__((ext_vector_type(16))) _Float16 v16h;
typedef __attribute__((ext_vector_type(8)))  _Float16 v8h;
typedef __attribute__((ext_vector_type(8)))  float    v8f;

#define H1      200
#define KPAD    384
#define TB      25600   // T*B = 100*256
#define NB      256     // batch
#define NT      100     // timesteps

#define USE_ASYNC_LDS 1

// ---------------------------------------------------------------------------
// Kernel 1: pack Wh_{L,M,R} (f32, [3,200,F]) into f16 buffer [9][200][384],
// zero-padded in K. Dendrite d: r = d/3 (0=L,1=M,2=R), c = d%3.
// ---------------------------------------------------------------------------
__global__ void wconv_kernel(const float* __restrict__ WL,
                             const float* __restrict__ WM,
                             const float* __restrict__ WR,
                             _Float16* __restrict__ wbf) {
    int idx = blockIdx.x * 256 + threadIdx.x;
    if (idx >= 9 * H1 * KPAD) return;
    int d   = idx / (H1 * KPAD);
    int rem = idx - d * (H1 * KPAD);
    int o   = rem / KPAD;
    int k   = rem - o * KPAD;
    int r   = d / 3, c = d - r * 3;
    float v = 0.0f;
    if (r == 0)      { if (k < 320) v = WL[(c * H1 + o) * 320 + k]; }
    else if (r == 1) {              v = WM[(c * H1 + o) * 384 + k]; }
    else             { if (k < 320) v = WR[(c * H1 + o) * 320 + k]; }
    wbf[idx] = (_Float16)v;
}

// ---------------------------------------------------------------------------
// Kernel 2: convert x f32 -> f16, region-packed: xpack[(m*3+c)*1024 + f'],
// f' = L-window(320) | M-window(384) | R-window(320). Makes the GEMM A-tile a
// contiguous, 16B-aligned byte copy (async-to-LDS friendly).
// ---------------------------------------------------------------------------
__global__ void xconv_kernel(const float* __restrict__ x,
                             _Float16* __restrict__ xp) {
    size_t idx = (size_t)blockIdx.x * 256 + threadIdx.x;
    if (idx >= (size_t)TB * 3 * 1024) return;
    unsigned pix = (unsigned)(idx & 1023);
    unsigned i = pix >> 5, j = pix & 31;
    size_t rest = idx >> 10;             // m*3 + c
    unsigned fp;
    if (j < 10)      fp = i * 10 + j;                  // L: cols 0..9
    else if (j < 22) fp = 320 + i * 12 + (j - 10);     // M: cols 10..21
    else             fp = 704 + i * 10 + (j - 22);     // R: cols 22..31
    xp[rest * 1024 + fp] = (_Float16)x[idx];
}

// ---------------------------------------------------------------------------
// Kernel 3: dendritic-current GEMM via v_wmma_f32_16x16x32_f16.
// Grid: (TB/128, 7, 9). Block: 256 threads = 8 waves; tile M=128 x N=32.
// Each wave: one 16-row strip, TWO accumulators (N halves) sharing the A frag.
// A staged in LDS via global_load_async_to_lds_b128 (ASYNCcnt).
// curbuf layout: [t][d][b][200] f32.
// ---------------------------------------------------------------------------
__global__ void gemm_kernel(const _Float16* __restrict__ xp,
                            const _Float16* __restrict__ wbf,
                            float* __restrict__ curbuf) {
    __shared__ _Float16 ldsA[128 * KPAD];   // 96 KB of 320 KB WGP LDS

    const int m0 = blockIdx.x * 128;
    const int n0 = blockIdx.y * 32;
    const int d  = blockIdx.z;
    const int r  = d / 3, c = d - r * 3;
    const int F    = (r == 1) ? 384 : 320;
    const int roff = (r == 0) ? 0 : ((r == 1) ? 320 : 704);

    // ---- stage A tile: 128 rows x F halves (raw copy, 16B chunks) ----
    const int cpr = F >> 3;                 // b128 chunks per row (40 or 48)
    const int nch = 128 * cpr;
#if USE_ASYNC_LDS
    // LDS aperture flat address: low 32 bits are the LDS byte offset.
    const unsigned ldsbase = (unsigned)(uintptr_t)(void*)&ldsA[0];
    for (int e = threadIdx.x; e < nch; e += 256) {
        int row = e / cpr;
        int cw  = e - row * cpr;
        unsigned loff = ldsbase + (unsigned)(row * KPAD + cw * 8) * 2u;
        unsigned long long ga = (unsigned long long)(uintptr_t)
            (xp + ((size_t)(m0 + row) * 3 + c) * 1024 + roff + cw * 8);
        asm volatile("global_load_async_to_lds_b128 %0, %1, off"
                     :: "v"(loff), "v"(ga) : "memory");
    }
    asm volatile("s_wait_asynccnt 0x0" ::: "memory");
#else
    for (int e = threadIdx.x; e < nch; e += 256) {
        int row = e / cpr;
        int cw  = e - row * cpr;
        const v8h* src = (const v8h*)
            (xp + ((size_t)(m0 + row) * 3 + c) * 1024 + roff + cw * 8);
        *(v8h*)&ldsA[row * KPAD + cw * 8] = *src;
    }
#endif
    __syncthreads();

    const int lane = threadIdx.x & 31;
    const int wm   = threadIdx.x >> 5;   // 0..7 (M strip)
    const int half = lane >> 4;          // hi/lo half-wave
    const int l16  = lane & 15;

    const int arow = wm * 16 + l16;      // A row for this lane (ISA A layout)
    const int col0 = n0 + l16;           // accumulator 0 column
    const int col1 = col0 + 16;          // accumulator 1 column
    const int c0 = col0 < H1 ? col0 : (H1 - 1);
    const int c1 = col1 < H1 ? col1 : (H1 - 1);
    const _Float16* __restrict__ wb0 = wbf + ((size_t)d * H1 + c0) * KPAD;
    const _Float16* __restrict__ wb1 = wbf + ((size_t)d * H1 + c1) * KPAD;

    v8f acc0 = {}, acc1 = {};
    for (int k0 = 0; k0 < F; k0 += 32) {
        // A fragment (contiguous k order, same as B): lane<16 -> K {0..15},
        // lane>=16 -> K {16..31}; two 16B LDS loads, reused by 2 WMMAs.
        const int ab = arow * KPAD + k0 + half * 16;
        v8h lo = *(const v8h*)&ldsA[ab];
        v8h hi = *(const v8h*)&ldsA[ab + 8];
        v16h a = __builtin_shufflevector(lo, hi, 0, 1, 2, 3, 4, 5, 6, 7,
                                                 8, 9, 10, 11, 12, 13, 14, 15);
        // B fragment (32x16): lanes 0-15 K k0..k0+15, lanes 16-31 K k0+16..31,
        // 16 contiguous halves per lane, 32B aligned.
        v16h b0 = *(const v16h*)&wb0[k0 + half * 16];
        v16h b1 = *(const v16h*)&wb1[k0 + half * 16];
        acc0 = __builtin_amdgcn_wmma_f32_16x16x32_f16(
            false, a, false, b0, (short)0, acc0, false, false);
        acc1 = __builtin_amdgcn_wmma_f32_16x16x32_f16(
            false, a, false, b1, (short)0, acc1, false, false);
    }

    // C layout: VGPR v -> row (v + 8*half), lane%16 -> col.
    const int mrow = m0 + wm * 16 + half * 8;
    if (col0 < H1) {
        #pragma unroll
        for (int v = 0; v < 8; ++v) {
            int m = mrow + v;                      // m = t*256 + b
            curbuf[(((size_t)(m >> 8) * 9 + d) * NB + (m & 255)) * H1 + col0] = acc0[v];
        }
    }
    if (col1 < H1) {
        #pragma unroll
        for (int v = 0; v < 8; ++v) {
            int m = mrow + v;
            curbuf[(((size_t)(m >> 8) * 9 + d) * NB + (m & 255)) * H1 + col1] = acc1[v];
        }
    }
}

// ---------------------------------------------------------------------------
// Kernel 4: LIF recurrence. One block per batch element, states in registers.
//   i' = 0.8*i + cur ;  v' = v + 0.1*(i' - v) ; spike if v' > 1 -> reset.
// ---------------------------------------------------------------------------
__global__ void rec_kernel(const float* __restrict__ curbuf,
                           const float* __restrict__ bh,
                           const float* __restrict__ Wo,
                           const float* __restrict__ bo,
                           float* __restrict__ out) {
    __shared__ float zsh[H1];
    __shared__ float sdo[40];
    __shared__ float WoS[4 * 10 * 50];

    const int b = blockIdx.x;
    const int h = threadIdx.x;            // 0..255

    for (int e = h; e < 2000; e += 256) WoS[e] = Wo[e];

    float bhd[9], id[9], vd[9];
    if (h < H1) {
        #pragma unroll
        for (int dd = 0; dd < 9; ++dd) {
            bhd[dd] = bh[dd * H1 + h];
            id[dd] = 0.0f; vd[dd] = 0.0f;
        }
    }
    float is_ = 0.0f, vs_ = 0.0f;                 // hidden soma
    const int ch = h / 10, o = h - ch * 10;       // valid for h < 40
    float ido = 0.0f, vdo = 0.0f;                 // out dendrite
    float bov = (h < 40) ? bo[h] : 0.0f;
    float iso = 0.0f, vso = 0.0f;                 // LI readout
    __syncthreads();

    for (int t = 0; t < NT; ++t) {
        if (h < H1) {
            float ssum = 0.0f;
            const float* cb = curbuf + (((size_t)t * 9) * NB + b) * H1 + h;
            #pragma unroll
            for (int dd = 0; dd < 9; ++dd) {
                float cur = cb[(size_t)dd * NB * H1] + bhd[dd];
                id[dd] = id[dd] * 0.8f + cur;
                vd[dd] = vd[dd] + 0.1f * (id[dd] - vd[dd]);
                float s = (vd[dd] > 1.0f) ? 1.0f : 0.0f;
                vd[dd] *= (1.0f - s);
                ssum += s;
            }
            is_ = is_ * 0.8f + ssum;
            vs_ = vs_ + 0.1f * (is_ - vs_);
            float z = (vs_ > 1.0f) ? 1.0f : 0.0f;
            vs_ *= (1.0f - z);
            zsh[h] = z;
        }
        __syncthreads();
        if (h < 40) {
            float co = bov;
            const float* wrow = &WoS[ch * 500 + o * 50];
            const float* zc   = &zsh[ch * 50];
            #pragma unroll 10
            for (int k = 0; k < 50; ++k) co += zc[k] * wrow[k];
            ido = ido * 0.8f + co;
            vdo = vdo + 0.1f * (ido - vdo);
            float s = (vdo > 1.0f) ? 1.0f : 0.0f;
            vdo *= (1.0f - s);
            sdo[h] = s;
        }
        __syncthreads();
        if (h < 10) {
            float ss = sdo[h] + sdo[10 + h] + sdo[20 + h] + sdo[30 + h];
            iso = iso * 0.8f + ss;
            vso = vso + 0.1f * (iso - vso);
            out[((size_t)t * NB + b) * 10 + h] = vso;
        }
    }
}

// ---------------------------------------------------------------------------
extern "C" void kernel_launch(void* const* d_in, const int* in_sizes, int n_in,
                              void* d_out, int out_size, void* d_ws, size_t ws_size,
                              hipStream_t stream) {
    const float* x  = (const float*)d_in[0];  // [100,256,3,32,32]
    const float* WL = (const float*)d_in[1];  // [3,200,320]
    const float* WM = (const float*)d_in[2];  // [3,200,384]
    const float* WR = (const float*)d_in[3];  // [3,200,320]
    const float* bh = (const float*)d_in[4];  // [9,200]
    const float* Wo = (const float*)d_in[5];  // [4,10,50]
    const float* bo = (const float*)d_in[6];  // [4,10]
    float* out = (float*)d_out;               // [100,256,10]

    const size_t wbf_bytes   = (size_t)9 * H1 * KPAD * 2;        //  1.35 MB
    const size_t xpack_bytes = (size_t)TB * 3 * 1024 * 2;        // 157.3 MB
    _Float16* wbf   = (_Float16*)d_ws;
    _Float16* xpack = (_Float16*)((char*)d_ws + wbf_bytes);
    float*    curbf = (float*)((char*)d_ws + wbf_bytes + xpack_bytes);

    {   // pack weights to f16
        int n = 9 * H1 * KPAD;
        wconv_kernel<<<(n + 255) / 256, 256, 0, stream>>>(WL, WM, WR, wbf);
    }
    {   // pack x to f16, region-major (async-LDS-friendly layout)
        size_t n = (size_t)TB * 3 * 1024;
        xconv_kernel<<<(unsigned)(n / 256), 256, 0, stream>>>(x, xpack);
    }
    {   // WMMA GEMM for all timesteps at once
        dim3 grid(TB / 128, (H1 + 31) / 32, 9);
        gemm_kernel<<<grid, 256, 0, stream>>>(xpack, wbf, curbf);
    }
    {   // sequential LIF recurrence
        rec_kernel<<<NB, 256, 0, stream>>>(curbf, bh, Wo, bo, out);
    }
}
